// RetBlock_31353261261569
// MI455X (gfx1250) — matrix operand order repaired and checked by reading, not compiled, not run on hardware
//
#include <hip/hip_runtime.h>
#include <hip/hip_bf16.h>

// ---------------------------------------------------------------------------
// CDNA5 (gfx1250) WMMA / TDM types
// ---------------------------------------------------------------------------
typedef __attribute__((ext_vector_type(16))) _Float16 v16h;
typedef __attribute__((ext_vector_type(8)))  float    v8f;
typedef __attribute__((ext_vector_type(4)))  unsigned int v4u;
typedef __attribute__((ext_vector_type(8)))  int      v8i;
typedef __attribute__((ext_vector_type(4)))  int      v4i;

#define B_   8
#define HH_  32
#define WW_  32
#define E_   256
#define L_   1024
#define HEADS_ 8
#define KD_  32
#define FFN_ 1024
#define NQ_  64
#define SCALING_ 0.17677669529663687f

#if defined(__has_builtin)
#  if __has_builtin(__builtin_amdgcn_tensor_load_to_lds)
#    define HAVE_TDM 1
#  endif
#endif
#ifndef HAVE_TDM
#  define HAVE_TDM 0
#endif

#if HAVE_TDM
// ---------------------------------------------------------------------------
// Tensor Data Mover: 2D tile load (global f32 -> LDS), ISA 08_async_tensor §8
// D# group0: [1:0]=count=1, [63:32]=lds_addr, [120:64]=global_addr,
//            [127:126]=type=2.  group1: [17:16]=data_size(2==4B),
//            [79:48]=tensor_dim0, [111:80]=tensor_dim1, [127:112]=tile_dim0,
//            [143:128]=tile_dim1, [207:160]=tensor_dim0_stride.
// ---------------------------------------------------------------------------
__device__ __forceinline__ void tdm_load_2d(unsigned lds_addr, const void* gptr,
                                            unsigned k_elems, unsigned rows_tot,
                                            unsigned tile_cols, unsigned tile_rows) {
  const unsigned long long ga = (unsigned long long)(size_t)gptr;
  v4u g0;
  g0[0] = 1u;                                     // count = 1 (valid)
  g0[1] = lds_addr;                               // LDS byte address
  g0[2] = (unsigned)(ga & 0xffffffffull);         // global_addr[31:0]
  g0[3] = (unsigned)((ga >> 32) & 0x01ffffffull)  // global_addr[56:32]
          | 0x80000000u;                          // type = 2 ("image")
  v8i g1;
  g1[0] = (int)(2u << 16);                        // data_size = 4 bytes
  g1[1] = (int)((k_elems & 0xffffu) << 16);       // tensor_dim0[15:0]
  g1[2] = (int)(((k_elems >> 16) & 0xffffu) | ((rows_tot & 0xffffu) << 16));
  g1[3] = (int)(((rows_tot >> 16) & 0xffffu) | (tile_cols << 16));
  g1[4] = (int)(tile_rows & 0xffffu);             // tile_dim1; tile_dim2 = 0
  g1[5] = (int)k_elems;                           // tensor_dim0_stride[31:0]
  g1[6] = 0;                                      // stride[47:32], dim1_stride
  g1[7] = 0;
  const v4i z4 = {0, 0, 0, 0};
#if __clang_major__ >= 23
  const v8i z8 = {0, 0, 0, 0, 0, 0, 0, 0};
  __builtin_amdgcn_tensor_load_to_lds(g0, g1, z4, z4, z8, 0);
#else
  __builtin_amdgcn_tensor_load_to_lds(g0, g1, z4, z4, 0);
#endif
}
#endif  // HAVE_TDM

// ---------------------------------------------------------------------------
// Depthwise conv (NHWC, per-channel KxK), optional residual add of the input
// ---------------------------------------------------------------------------
__global__ void dwconv_kernel(const float* __restrict__ in,
                              const float* __restrict__ w,
                              const float* __restrict__ bias,
                              float* __restrict__ out,
                              int ksz, int pad, int add_input) {
  const int idx = blockIdx.x * blockDim.x + threadIdx.x;
  if (idx >= B_ * HH_ * WW_ * E_) return;
  const int e  = idx & (E_ - 1);
  const int xw = (idx >> 8) & (WW_ - 1);
  const int y  = (idx >> 13) & (HH_ - 1);
  const int b  = idx >> 18;
  float acc = bias[e];
  for (int ky = 0; ky < ksz; ++ky) {
    const int iy = y + ky - pad;
    if (iy < 0 || iy >= HH_) continue;
    for (int kx = 0; kx < ksz; ++kx) {
      const int ix = xw + kx - pad;
      if (ix < 0 || ix >= WW_) continue;
      acc += in[(((b * HH_ + iy) * WW_ + ix) << 8) | e] *
             w[(e * ksz + ky) * ksz + kx];
    }
  }
  if (add_input) acc += in[idx];
  out[idx] = acc;
}

// ---------------------------------------------------------------------------
// LayerNorm over last dim == 256; one wave (32 lanes) per row, 8 elems/lane
// ---------------------------------------------------------------------------
__global__ void layernorm256_kernel(const float* __restrict__ in,
                                    const float* __restrict__ g,
                                    const float* __restrict__ bta,
                                    float* __restrict__ out, int rows) {
  const int wid = threadIdx.x >> 5, lane = threadIdx.x & 31;
  const int row = blockIdx.x * 8 + wid;
  if (row >= rows) return;
  const float* p = in + (size_t)row * 256;
  float vals[8];
  float s = 0.f;
#pragma unroll
  for (int j = 0; j < 8; ++j) { vals[j] = p[j * 32 + lane]; s += vals[j]; }
#pragma unroll
  for (int off = 16; off; off >>= 1) s += __shfl_xor(s, off, 32);
  const float mean = s * (1.0f / 256.0f);
  float vsum = 0.f;
#pragma unroll
  for (int j = 0; j < 8; ++j) { const float d = vals[j] - mean; vsum += d * d; }
#pragma unroll
  for (int off = 16; off; off >>= 1) vsum += __shfl_xor(vsum, off, 32);
  const float inv = rsqrtf(vsum * (1.0f / 256.0f) + 1e-6f);
  float* o = out + (size_t)row * 256;
#pragma unroll
  for (int j = 0; j < 8; ++j) {
    const int e = j * 32 + lane;
    o[e] = (vals[j] - mean) * inv * g[e] + bta[e];
  }
}

// ---------------------------------------------------------------------------
// WMMA GEMM with TDM-staged operand panels.
//   C[M,N] = act((A[M,K] . Wt[N,K]^T + bias) * scale) + residual
// Block: 256 threads (8 waves), tile 64x64, K chunked by KC=256.
// LDS: A panel 64xKC f32 (64KB) + B panel 64xKC f32 (64KB) = 128KB dynamic,
// each filled by ONE tensor_load_to_lds descriptor (TENSORcnt-tracked).
// Each wave computes a 16x32 strip: 2 v_wmma_f32_16x16x32_f16 per k-step.
// ---------------------------------------------------------------------------
#define KC_ 256
__global__ void gemm_wmma_kernel(const float* __restrict__ A,
                                 const float* __restrict__ Wt,
                                 const float* __restrict__ bias,
                                 const float* __restrict__ residual,
                                 float* __restrict__ C,
                                 int M, int N, int K, float scale, int act) {
  extern __shared__ char gsm[];
  float* AsF = (float*)gsm;            // 64 x KC_
  float* BsF = AsF + 64 * KC_;         // 64 x KC_  (rows of Wt: [n][k])
  const int tid = threadIdx.x;
  const int wid = tid >> 5, lane = tid & 31;
  const int n0 = blockIdx.x << 6, m0 = blockIdx.y << 6;
  const int wr = wid >> 1, wc = wid & 1;
  const int m = lane & 15, n = lane & 15;
  const int kb8 = (lane >> 4) << 3, kb16 = (lane >> 4) << 4;
  const int rbase = (lane >> 4) << 3;
  v8f acc0 = {}, acc1 = {};

  for (int kc = 0; kc < K; kc += KC_) {
    __syncthreads();  // previous chunk fully consumed before overwrite
#if HAVE_TDM
    if (tid < 32) {   // wave 0 drives the Tensor Data Mover
      tdm_load_2d((unsigned)(size_t)AsF, &A[(size_t)m0 * K + kc], (unsigned)K,
                  (unsigned)M, KC_, 64);
      tdm_load_2d((unsigned)(size_t)BsF, &Wt[(size_t)n0 * K + kc], (unsigned)K,
                  (unsigned)N, KC_, 64);
      __builtin_amdgcn_s_wait_tensorcnt(0);
    }
#else
    for (int idx = tid; idx < 64 * (KC_ / 4); idx += 256) {
      const int r = idx >> 6, c4 = (idx & 63) << 2;
      *(float4*)&AsF[r * KC_ + c4] =
          *(const float4*)&A[(size_t)(m0 + r) * K + kc + c4];
      *(float4*)&BsF[r * KC_ + c4] =
          *(const float4*)&Wt[(size_t)(n0 + r) * K + kc + c4];
    }
#endif
    if (kc + KC_ < K)  // gfx1250 global_prefetch_b8 of next A chunk
      __builtin_prefetch(&A[(size_t)(m0 + (tid & 63)) * K + kc + KC_], 0, 3);
    __syncthreads();

    for (int k0 = 0; k0 < KC_; k0 += 32) {
      v16h af, b0, b1;
#pragma unroll
      for (int i = 0; i < 16; ++i) {
        const int kk = kb16 + i;
        af[i] = (_Float16)AsF[(wr * 16 + m) * KC_ + k0 + kk];
      }
#pragma unroll
      for (int i = 0; i < 16; ++i) {
        b0[i] = (_Float16)BsF[(wc * 32 + n) * KC_ + k0 + kb16 + i];
        b1[i] = (_Float16)BsF[(wc * 32 + 16 + n) * KC_ + k0 + kb16 + i];
      }
      acc0 = __builtin_amdgcn_wmma_f32_16x16x32_f16(false, af, false, b0,
                                                    (short)0, acc0, false, false);
      acc1 = __builtin_amdgcn_wmma_f32_16x16x32_f16(false, af, false, b1,
                                                    (short)0, acc1, false, false);
    }
  }

  const int col0 = n0 + wc * 32 + n;
  const int col1 = col0 + 16;
  const float bc0 = bias ? bias[col0] : 0.f;
  const float bc1 = bias ? bias[col1] : 0.f;
#pragma unroll
  for (int r = 0; r < 8; ++r) {
    const int row = m0 + wr * 16 + rbase + r;
    float v0 = (acc0[r] + bc0) * scale;
    float v1 = (acc1[r] + bc1) * scale;
    if (act == 1) {
      v0 = 0.5f * v0 * (1.0f + erff(v0 * 0.70710678118654752f));
      v1 = 0.5f * v1 * (1.0f + erff(v1 * 0.70710678118654752f));
    }
    if (residual) {
      v0 += residual[(size_t)row * N + col0];
      v1 += residual[(size_t)row * N + col1];
    }
    C[(size_t)row * N + col0] = v0;
    C[(size_t)row * N + col1] = v1;
  }
}

// ---------------------------------------------------------------------------
// theta_shift (rotary-style) applied in place to q and k.
// ---------------------------------------------------------------------------
__global__ void theta_shift_kernel(float* __restrict__ q, float* __restrict__ k,
                                   const float* __restrict__ sinp,
                                   const float* __restrict__ cosp,
                                   int total_pairs) {
  const int p = blockIdx.x * blockDim.x + threadIdx.x;
  if (p >= total_pairs) return;
  const int e0 = p << 1;
  const int l = (e0 >> 8) & (L_ - 1);
  const int d = e0 & (KD_ - 1);
  const float c0 = cosp[l * KD_ + d], c1 = cosp[l * KD_ + d + 1];
  const float s0 = sinp[l * KD_ + d], s1 = sinp[l * KD_ + d + 1];
  const float q0 = q[e0], q1 = q[e0 + 1];
  q[e0]     = q0 * c0 - q1 * s0;
  q[e0 + 1] = q1 * c1 + q0 * s1;
  const float k0 = k[e0], k1 = k[e0 + 1];
  k[e0]     = k0 * c0 - k1 * s0;
  k[e0 + 1] = k1 * c1 + k0 * s1;
}

// ---------------------------------------------------------------------------
// Fused WMMA attention: one block = one (b, head, 16-row q tile).
// 256 threads (8 waves). Dyn LDS: 16*Lk f32 scores + 16*Lk f16 probs = 96KB.
// ---------------------------------------------------------------------------
__global__ void attention_wmma_kernel(
    const float* __restrict__ Qb, long long q_bs, int q_ts,
    const float* __restrict__ Kb, long long k_bs, int k_ts,
    const float* __restrict__ Vb, long long v_bs, int v_ts,
    const float* __restrict__ mask,   // (HEADS, Lk, Lk) or null
    const float* __restrict__ extra,  // same layout as Out, or null
    float* __restrict__ Out, long long o_bs, int o_ts,
    int Lk, float scale) {
  extern __shared__ char smem[];
  float* scores = (float*)smem;                                    // 16*Lk f32
  _Float16* probs = (_Float16*)(smem + (size_t)16 * Lk * sizeof(float));
  __shared__ float rowsum[16];

  const int tid = threadIdx.x;
  const int wid = tid >> 5, lane = tid & 31;
  const int q0 = blockIdx.x << 4;
  const int h = blockIdx.y;
  const int b = blockIdx.z;

  const float* qp = Qb + (size_t)b * q_bs + (size_t)q0 * q_ts + h * KD_;
  const float* kbase = Kb + (size_t)b * k_bs + h * KD_;
  const float* vbase = Vb + (size_t)b * v_bs + h * KD_;

  const int m = lane & 15;
  const int n = lane & 15;
  const int kb8 = (lane >> 4) << 3;
  const int kb16 = (lane >> 4) << 4;
  const int rbase = (lane >> 4) << 3;

  v16h aq;
#pragma unroll
  for (int i = 0; i < 16; ++i) {
    const int k = kb16 + i;
    aq[i] = (_Float16)qp[m * q_ts + k];
  }

  // ---- phase 1: score tiles S(16x16) = Q(16x32) . K_tile^T(32x16) ----
  for (int t = wid; t < (Lk >> 4); t += 8) {
    const float* kp = kbase + (size_t)(t << 4) * k_ts;
    v16h bf;
#pragma unroll
    for (int i = 0; i < 16; ++i) bf[i] = (_Float16)kp[n * k_ts + kb16 + i];
    v8f s = {};
    s = __builtin_amdgcn_wmma_f32_16x16x32_f16(false, aq, false, bf,
                                               (short)0, s, false, false);
    const int col = (t << 4) + n;
#pragma unroll
    for (int r = 0; r < 8; ++r) {
      const int row = rbase + r;
      float val = s[r] * scale;
      if (mask) val += mask[((size_t)h * Lk + (q0 + row)) * Lk + col];
      scores[row * Lk + col] = val;
    }
  }
  __syncthreads();

  // ---- phase 2: row softmax ----
  for (int rr = wid; rr < 16; rr += 8) {
    const float* srow = scores + rr * Lk;
    float mx = -3.0e38f;
    for (int c = lane; c < Lk; c += 32) mx = fmaxf(mx, srow[c]);
#pragma unroll
    for (int off = 16; off; off >>= 1) mx = fmaxf(mx, __shfl_xor(mx, off, 32));
    float sum = 0.f;
    for (int c = lane; c < Lk; c += 32) {
      const float e = __expf(srow[c] - mx);
      probs[rr * Lk + c] = (_Float16)e;
      sum += e;
    }
#pragma unroll
    for (int off = 16; off; off >>= 1) sum += __shfl_xor(sum, off, 32);
    if (lane == 0) rowsum[rr] = sum;
  }
  __syncthreads();

  // ---- phase 3: out(16x32) = P(16xLk) . V(Lk x 32) ----
  v8f acc0 = {}, acc1 = {};
  for (int t = wid; t < (Lk >> 5); t += 8) {
    v16h ap;
#pragma unroll
    for (int i = 0; i < 16; ++i) {
      const int k = kb16 + i;
      ap[i] = probs[m * Lk + (t << 5) + k];
    }
    const float* vp = vbase + (size_t)(t << 5) * v_ts;
    v16h bv0, bv1;
#pragma unroll
    for (int i = 0; i < 16; ++i) {
      bv0[i] = (_Float16)vp[(kb16 + i) * v_ts + n];
      bv1[i] = (_Float16)vp[(kb16 + i) * v_ts + 16 + n];
    }
    acc0 = __builtin_amdgcn_wmma_f32_16x16x32_f16(false, ap, false, bv0,
                                                  (short)0, acc0, false, false);
    acc1 = __builtin_amdgcn_wmma_f32_16x16x32_f16(false, ap, false, bv1,
                                                  (short)0, acc1, false, false);
  }
  float* part = scores;  // reuse score region: 8 waves * 16 rows * 32 cols
#pragma unroll
  for (int r = 0; r < 8; ++r) {
    const int row = rbase + r;
    part[(wid * 16 + row) * 32 + n] = acc0[r];
    part[(wid * 16 + row) * 32 + 16 + n] = acc1[r];
  }
  __syncthreads();

  // ---- phase 4: cross-wave reduce, normalize, add extra, store ----
  for (int idx = tid; idx < 512; idx += 256) {
    const int row = idx >> 5, col = idx & 31;
    float s = 0.f;
#pragma unroll
    for (int w = 0; w < 8; ++w) s += part[(w * 16 + row) * 32 + col];
    s /= rowsum[row];
    const size_t off =
        (size_t)b * o_bs + (size_t)(q0 + row) * o_ts + h * KD_ + col;
    if (extra) s += extra[off];
    Out[off] = s;
  }
}

// ---------------------------------------------------------------------------
// Host orchestration
// ---------------------------------------------------------------------------
extern "C" void kernel_launch(void* const* d_in, const int* in_sizes, int n_in,
                              void* d_out, int out_size, void* d_ws,
                              size_t ws_size, hipStream_t stream) {
  (void)in_sizes; (void)n_in; (void)out_size; (void)ws_size;
  const float* x       = (const float*)d_in[0];
  const float* queries = (const float*)d_in[1];
  const float* sinp    = (const float*)d_in[2];
  const float* cosp    = (const float*)d_in[3];
  const float* mask    = (const float*)d_in[4];
  const float* ln1_g   = (const float*)d_in[5];
  const float* ln1_b   = (const float*)d_in[6];
  const float* ln2_g   = (const float*)d_in[7];
  const float* ln2_b   = (const float*)d_in[8];
  const float* wq = (const float*)d_in[9];   const float* bq = (const float*)d_in[10];
  const float* wk = (const float*)d_in[11];  const float* bk = (const float*)d_in[12];
  const float* wv = (const float*)d_in[13];  const float* bv = (const float*)d_in[14];
  const float* lepe_w = (const float*)d_in[15]; const float* lepe_b = (const float*)d_in[16];
  const float* wo = (const float*)d_in[17];  const float* bo = (const float*)d_in[18];
  const float* fc1_w = (const float*)d_in[19]; const float* fc1_b = (const float*)d_in[20];
  const float* fc2_w = (const float*)d_in[21]; const float* fc2_b = (const float*)d_in[22];
  const float* pos_w = (const float*)d_in[23]; const float* pos_b = (const float*)d_in[24];
  const float* ca_in_w  = (const float*)d_in[25]; const float* ca_in_b  = (const float*)d_in[26];
  const float* ca_out_w = (const float*)d_in[27]; const float* ca_out_b = (const float*)d_in[28];

  const size_t SZX = (size_t)B_ * L_ * E_;  // 2,097,152
  float* ws   = (float*)d_ws;
  float* X1   = ws;
  float* HN   = X1 + SZX;
  float* Qb   = HN + SZX;
  float* Kb   = Qb + SZX;
  float* Vb   = Kb + SZX;
  float* LEPE = Vb + SZX;
  float* AOUT = LEPE + SZX;
  float* X2   = AOUT + SZX;
  float* BIG  = X2 + SZX;          // 8192x1024; reused for KV / QH1
  float* QLN  = BIG + (size_t)8192 * 1024;
  float* QP   = QLN + (size_t)512 * 256;
  float* CAO  = QP + (size_t)512 * 256;
  float* Q2   = CAO + (size_t)512 * 256;
  float* Q2LN = Q2 + (size_t)512 * 256;
  float* KV   = BIG;                        // 8192 x 512
  float* QH1  = BIG + (size_t)8192 * 512;   // 512 x 1024

  float* xout = (float*)d_out;
  float* qout = xout + SZX;

  const int attn_lds = 16 * 1024 * (int)(sizeof(float) + sizeof(_Float16));
  const int gemm_lds = 2 * 64 * KC_ * (int)sizeof(float);  // 128 KB
  (void)hipFuncSetAttribute(
      reinterpret_cast<const void*>(&attention_wmma_kernel),
      hipFuncAttributeMaxDynamicSharedMemorySize, attn_lds);
  (void)hipFuncSetAttribute(
      reinterpret_cast<const void*>(&gemm_wmma_kernel),
      hipFuncAttributeMaxDynamicSharedMemorySize, gemm_lds);

  // 1) x = x + dwconv3x3(x)
  dwconv_kernel<<<8192, 256, 0, stream>>>(x, pos_w, pos_b, X1, 3, 1, 1);
  // 2) hn = LN1(x)
  layernorm256_kernel<<<1024, 256, 0, stream>>>(X1, ln1_g, ln1_b, HN, 8192);
  // 3) q, k (*SCALING), v projections (M=8192, N=256, K=256)
  gemm_wmma_kernel<<<dim3(4, 128), 256, gemm_lds, stream>>>(HN, wq, bq, nullptr, Qb, 8192, 256, 256, 1.f, 0);
  gemm_wmma_kernel<<<dim3(4, 128), 256, gemm_lds, stream>>>(HN, wk, bk, nullptr, Kb, 8192, 256, 256, SCALING_, 0);
  gemm_wmma_kernel<<<dim3(4, 128), 256, gemm_lds, stream>>>(HN, wv, bv, nullptr, Vb, 8192, 256, 256, 1.f, 0);
  // 4) lepe = dwconv5x5(v)
  dwconv_kernel<<<8192, 256, 0, stream>>>(Vb, lepe_w, lepe_b, LEPE, 5, 2, 0);
  // 5) theta shift q, k in place
  theta_shift_kernel<<<4096, 256, 0, stream>>>(Qb, Kb, sinp, cosp, (int)(SZX / 2));
  // 6) retention attention (+mask, +lepe)
  attention_wmma_kernel<<<dim3(64, HEADS_, B_), 256, attn_lds, stream>>>(
      Qb, (long long)(L_ * E_), E_, Kb, (long long)(L_ * E_), E_,
      Vb, (long long)(L_ * E_), E_, mask, LEPE,
      AOUT, (long long)(L_ * E_), E_, L_, 1.0f);
  // 7) x = x + out @ wo^T + bo
  gemm_wmma_kernel<<<dim3(4, 128), 256, gemm_lds, stream>>>(AOUT, wo, bo, X1, X2, 8192, 256, 256, 1.f, 0);
  // 8) x = x + FFN(LN2(x))
  layernorm256_kernel<<<1024, 256, 0, stream>>>(X2, ln2_g, ln2_b, HN, 8192);
  gemm_wmma_kernel<<<dim3(16, 128), 256, gemm_lds, stream>>>(HN, fc1_w, fc1_b, nullptr, BIG, 8192, 1024, 256, 1.f, 1);
  gemm_wmma_kernel<<<dim3(4, 128), 256, gemm_lds, stream>>>(BIG, fc2_w, fc2_b, X2, xout, 8192, 256, 1024, 1.f, 0);
  // 9) cross attention
  layernorm256_kernel<<<64, 256, 0, stream>>>(queries, ln1_g, ln1_b, QLN, 512);
  gemm_wmma_kernel<<<dim3(4, 8), 256, gemm_lds, stream>>>(QLN, ca_in_w, ca_in_b, nullptr, QP, 512, 256, 256, 1.f, 0);
  gemm_wmma_kernel<<<dim3(8, 128), 256, gemm_lds, stream>>>(xout, ca_in_w + 256 * 256, ca_in_b + 256, nullptr, KV, 8192, 512, 256, 1.f, 0);
  attention_wmma_kernel<<<dim3(4, HEADS_, B_), 256, attn_lds, stream>>>(
      QP, (long long)(NQ_ * E_), E_, KV, (long long)(L_ * 512), 512,
      KV + 256, (long long)(L_ * 512), 512, nullptr, nullptr,
      CAO, (long long)(NQ_ * E_), E_, L_, SCALING_);
  gemm_wmma_kernel<<<dim3(4, 8), 256, gemm_lds, stream>>>(CAO, ca_out_w, ca_out_b, queries, Q2, 512, 256, 256, 1.f, 0);
  // 10) queries = queries + FFN(LN2(queries))
  layernorm256_kernel<<<64, 256, 0, stream>>>(Q2, ln2_g, ln2_b, Q2LN, 512);
  gemm_wmma_kernel<<<dim3(16, 8), 256, gemm_lds, stream>>>(Q2LN, fc1_w, fc1_b, nullptr, QH1, 512, 1024, 256, 1.f, 1);
  gemm_wmma_kernel<<<dim3(4, 8), 256, gemm_lds, stream>>>(QH1, fc2_w, fc2_b, Q2, qout, 512, 256, 1024, 1.f, 0);
}
